// Generator_79748952752674
// MI455X (gfx1250) — matrix operand order repaired and checked by reading, not compiled, not run on hardware
//
#include <hip/hip_runtime.h>
#include <hip/hip_bf16.h>
#include <math.h>

typedef __attribute__((ext_vector_type(16))) _Float16 v16h;
typedef __attribute__((ext_vector_type(8)))  _Float16 v8h;
typedef __attribute__((ext_vector_type(8)))  float    v8f;
typedef __attribute__((ext_vector_type(4)))  unsigned int u32x4;
typedef __attribute__((ext_vector_type(8)))  unsigned int u32x8;

#define B_TOTAL 32768
#define M_TOP   200
#define DZV     20
#define D_INF   640     // 3*M + 2*DZ
#define CHN     96
#define THRESHV 0.1f

// Packed f16 weight sizes (halves)
#define W1_HALVES (D_INF * CHN)          // 61440 (120 frags * 512)
#define W2_HALVES (CHN * CHN)            // 9216  (18 frags)
#define W3_HALVES (CHN * CHN)            // 9216  (18 frags)
#define W4_HALVES (CHN * 16)             // 1536  (3 frags)
#define WLDS_HALVES (W2_HALVES + W3_HALVES + W4_HALVES)   // 19968 halves = 39936 B
#define WLDS_BYTES  (WLDS_HALVES * 2)

union AFrag { v16h v; v8h h[2]; };

__device__ __forceinline__ float gelu_tanh(float x) {
    // jax.nn.gelu (approximate=True)
    float x3 = x * x * x;
    return 0.5f * x * (1.0f + tanhf(0.7978845608028654f * (x + 0.044715f * x3)));
}

// ---------------------------------------------------------------------------
// Pack f32 W[K][N] (row-major, K multiple of 32) into f16 WMMA B-fragments.
// B layout (32x16 KxN): lane = N (mod 16); lanes 0-15 hold K=0..7,16..23,
// lanes 16-31 hold K=8..15,24..31 (half j<8: K=8h+j, j>=8: K=16+8h+j-8).
// ---------------------------------------------------------------------------
__global__ void pack_kernel(const float* __restrict__ W, _Float16* __restrict__ out,
                            int N, int NT, int KT) {
    int idx = blockIdx.x * blockDim.x + threadIdx.x;
    int total = NT * KT * 32;
    if (idx >= total) return;
    int f = idx >> 5;
    int l = idx & 31;
    int nt = f / KT;
    int kt = f - nt * KT;
    int n  = nt * 16 + (l & 15);
    int kbase = kt * 32 + ((l & 16) ? 8 : 0);
    _Float16* dst = out + ((size_t)(f * 32 + l)) * 16;
    #pragma unroll
    for (int j = 0; j < 16; ++j) {
        float v = (n < N) ? W[(size_t)(kbase + j + ((j >= 8) ? 8 : 0)) * N + n] : 0.0f;
        dst[j] = (_Float16)v;
    }
}

// ---------------------------------------------------------------------------
// Feature extraction, histogram-accelerated stable top-200 selection.
// Phase 1: 256-bucket histogram of intensities; find threshold bucket tb such
//          that strictly-higher buckets hold < 200 elements but adding tb's
//          crosses 200. Values in buckets >= tb are strictly greater than all
//          values in lower buckets, so ranking candidates among themselves
//          reproduces the global stable descending argsort ranks.
// Phase 2: O(ncand^2) exact rank (ncand ~ 205) with jnp.argsort tie-breaking.
// ---------------------------------------------------------------------------
__global__ void feat_kernel(const float* __restrict__ images,
                            const float* __restrict__ angles,
                            _Float16* __restrict__ X) {
    __shared__ float s_val[784];
    __shared__ int   s_hist[256];
    __shared__ float c_val[784];
    __shared__ short c_idx[784];
    __shared__ int   s_ncand;
    __shared__ int   s_tb;

    const int b   = blockIdx.x;
    const int tid = threadIdx.x;
    const float* img = images + (size_t)b * 784;

    for (int p = tid; p < 784; p += 256) s_val[p] = img[p];
    s_hist[tid] = 0;
    if (tid == 0) s_ncand = 0;
    __syncthreads();

    for (int p = tid; p < 784; p += 256) {
        int bkt = (int)(s_val[p] * 256.0f);
        bkt = bkt < 0 ? 0 : (bkt > 255 ? 255 : bkt);
        atomicAdd(&s_hist[bkt], 1);
    }
    __syncthreads();

    {   // per-bucket suffix count; unique crossing bucket writes s_tb
        int g = 0;
        for (int v = tid + 1; v < 256; ++v) g += s_hist[v];
        if (g < M_TOP && g + s_hist[tid] >= M_TOP) s_tb = tid;
    }
    __syncthreads();

    const int tb = s_tb;
    for (int p = tid; p < 784; p += 256) {
        float v = s_val[p];
        int bkt = (int)(v * 256.0f);
        bkt = bkt < 0 ? 0 : (bkt > 255 ? 255 : bkt);
        if (bkt >= tb) {
            int slot = atomicAdd(&s_ncand, 1);
            c_val[slot] = v;
            c_idx[slot] = (short)p;
        }
    }
    __syncthreads();

    _Float16* xrow = X + (size_t)b * D_INF;
    const int nc = s_ncand;
    for (int i = tid; i < nc; i += 256) {
        float vi = c_val[i];
        int   pi = c_idx[i];
        int cnt = 0;
        #pragma unroll 4
        for (int j = 0; j < nc; ++j) {
            float vj = c_val[j];
            int   pj = c_idx[j];
            cnt += (vj > vi) || ((vj == vi) && (pj < pi));
        }
        if (cnt < M_TOP) {
            int r = pi / 28, c = pi - r * 28;
            float cx = (c < 14) ? (float)(c - 14) : (float)(c - 13);
            float cy = (r < 14) ? (float)(14 - r) : (float)(13 - r);
            float in_ = vi;
            if (in_ < THRESHV) { cx = 0.0f; cy = 0.0f; in_ = 0.0f; }
            xrow[cnt]                 = (_Float16)in_;
            xrow[M_TOP + 2 * cnt]     = (_Float16)cx;
            xrow[M_TOP + 2 * cnt + 1] = (_Float16)cy;
        }
    }

    if (tid < DZV) {
        float a = angles[(size_t)b * DZV + tid];
        xrow[3 * M_TOP + 2 * tid]     = (_Float16)cosf(a);
        xrow[3 * M_TOP + 2 * tid + 1] = (_Float16)sinf(a);
    }
}

// ---------------------------------------------------------------------------
// Fused 4-layer MLP via v_wmma_f32_16x16x32_f16.
// 128 threads = 4 waves; each wave owns 16 batch rows. Layers 2-4 weights are
// DMA'd once per block into LDS by the Tensor Data Mover (tensor_load_to_lds),
// waited with s_wait_tensorcnt. B-fragments are hoisted per K-tile so load
// clauses overlap the WMMA stream.
// ---------------------------------------------------------------------------
struct SMEM {
    _Float16 wbuf[WLDS_HALVES];      // W2|W3|W4 fragments (39936 B), TDM target
    _Float16 hbuf[4][16][104];       // per-wave activation buffer (13312 B)
};

__global__ void __launch_bounds__(128) mlp_kernel(
        const _Float16* __restrict__ X,
        const _Float16* __restrict__ W1p, const _Float16* __restrict__ Wlds_src,
        const float* __restrict__ b1, const float* __restrict__ b2,
        const float* __restrict__ b3, const float* __restrict__ b4,
        float* __restrict__ out) {
    __shared__ SMEM sm;

    const int tid  = threadIdx.x;
    const int wave = tid >> 5;
    const int lane = tid & 31;
    const int rowBase = (blockIdx.x * 4 + wave) * 16;
    const int rsel = lane & 15;
    const int msk8 = (lane & 16) ? 8 : 0;

    // --- TDM: DMA 39936 B of packed W2|W3|W4 into LDS (one wave issues) ----
    if (wave == 0) {
        unsigned long long ga = (unsigned long long)(size_t)Wlds_src;
        unsigned int lds_off  = (unsigned int)(size_t)&sm.wbuf[0];
        const unsigned int n8 = WLDS_BYTES / 8;   // 4992 8-byte elements
        u32x4 g0;
        g0[0] = 1u;                                // count=1, user descriptor
        g0[1] = lds_off;                           // lds_addr (bytes)
        g0[2] = (unsigned int)ga;                  // global_addr[31:0]
        g0[3] = (unsigned int)(ga >> 32) | (2u << 30); // addr[56:32] | type=2
        u32x8 g1;
        g1[0] = (3u << 16);                        // data_size = 8 bytes
        g1[1] = (n8 & 0xFFFFu) << 16;              // tensor_dim0[15:0]
        g1[2] = (n8 >> 16) | (1u << 16);           // tensor_dim0[31:16] | tensor_dim1=1
        g1[3] = (n8 & 0xFFFFu) << 16;              // tile_dim0 = n8
        g1[4] = 0;                                 // tile_dim1/2 unused (1-D)
        g1[5] = n8;                                // tensor_dim0_stride[31:0]
        g1[6] = 0;
        g1[7] = 0;
        asm volatile("tensor_load_to_lds %0, %1" :: "s"(g0), "s"(g1) : "memory");
        __builtin_amdgcn_s_wait_tensorcnt(0);
    }
    __syncthreads();

    const _Float16* w2l = sm.wbuf;
    const _Float16* w3l = sm.wbuf + W2_HALVES;
    const _Float16* w4l = sm.wbuf + W2_HALVES + W3_HALVES;
    _Float16 (*hb)[104] = sm.hbuf[wave];

    const v8f zero = {0.f, 0.f, 0.f, 0.f, 0.f, 0.f, 0.f, 0.f};
    v8f acc[6];

    // ---------------- Layer 1: X(16x640) @ W1(640x96), W1 from global ------
    #pragma unroll
    for (int nt = 0; nt < 6; ++nt) acc[nt] = zero;

    for (int kt = 0; kt < 20; ++kt) {
        const _Float16* xp = X + (size_t)(rowBase + rsel) * D_INF + kt * 32 + msk8;
        AFrag a;
        a.h[0] = *(const v8h*)xp;
        a.h[1] = *(const v8h*)(xp + 16);
        __builtin_prefetch(xp + 32, 0, 3);
        AFrag bf[6];
        #pragma unroll
        for (int nt = 0; nt < 6; ++nt) {   // one clause of 12 b128 loads
            const _Float16* wp = W1p + ((size_t)((nt * 20 + kt) * 32 + lane)) * 16;
            bf[nt].h[0] = *(const v8h*)wp;
            bf[nt].h[1] = *(const v8h*)(wp + 8);
        }
        #pragma unroll
        for (int nt = 0; nt < 6; ++nt) {
            acc[nt] = __builtin_amdgcn_wmma_f32_16x16x32_f16(
                false, a.v, false, bf[nt].v, (short)0, acc[nt], false, false);
        }
    }
    #pragma unroll
    for (int nt = 0; nt < 6; ++nt) {
        float bias = b1[nt * 16 + rsel];
        #pragma unroll
        for (int i = 0; i < 8; ++i) {
            float v = acc[nt][i] + bias;
            hb[i + msk8][nt * 16 + rsel] = (_Float16)gelu_tanh(v);
        }
    }
    __syncthreads();

    // ---------------- Layers 2 & 3: h @ W (weights from LDS) ---------------
    const _Float16* Wh[2] = { w2l, w3l };
    const float*    bh[2] = { b2,  b3  };
    #pragma unroll 1
    for (int layer = 0; layer < 2; ++layer) {
        #pragma unroll
        for (int nt = 0; nt < 6; ++nt) acc[nt] = zero;
        #pragma unroll
        for (int kt = 0; kt < 3; ++kt) {
            const _Float16* ap = &hb[rsel][kt * 32 + msk8];
            AFrag a;
            a.h[0] = *(const v8h*)ap;
            a.h[1] = *(const v8h*)(ap + 16);
            AFrag bf[6];
            #pragma unroll
            for (int nt = 0; nt < 6; ++nt) {
                const _Float16* wp = Wh[layer] + ((size_t)((nt * 3 + kt) * 32 + lane)) * 16;
                bf[nt].h[0] = *(const v8h*)wp;
                bf[nt].h[1] = *(const v8h*)(wp + 8);
            }
            #pragma unroll
            for (int nt = 0; nt < 6; ++nt) {
                acc[nt] = __builtin_amdgcn_wmma_f32_16x16x32_f16(
                    false, a.v, false, bf[nt].v, (short)0, acc[nt], false, false);
            }
        }
        __syncthreads();
        #pragma unroll
        for (int nt = 0; nt < 6; ++nt) {
            float bias = bh[layer][nt * 16 + rsel];
            #pragma unroll
            for (int i = 0; i < 8; ++i) {
                float v = acc[nt][i] + bias;
                hb[i + msk8][nt * 16 + rsel] = (_Float16)gelu_tanh(v);
            }
        }
        __syncthreads();
    }

    // ---------------- Layer 4: h @ W4 (N padded to 16, weights in LDS) -----
    v8f acc4 = zero;
    #pragma unroll
    for (int kt = 0; kt < 3; ++kt) {
        const _Float16* ap = &hb[rsel][kt * 32 + msk8];
        AFrag a;
        a.h[0] = *(const v8h*)ap;
        a.h[1] = *(const v8h*)(ap + 16);
        const _Float16* wp = w4l + ((size_t)(kt * 32 + lane)) * 16;
        AFrag bf;
        bf.h[0] = *(const v8h*)wp;
        bf.h[1] = *(const v8h*)(wp + 8);
        acc4 = __builtin_amdgcn_wmma_f32_16x16x32_f16(
            false, a.v, false, bf.v, (short)0, acc4, false, false);
    }

    // out.reshape(B,2,2).transpose(0,2,1): swap linear elements 1 <-> 2
    if (rsel < 4) {
        int col = (rsel == 1) ? 2 : ((rsel == 2) ? 1 : rsel);
        float bias = b4[rsel];
        #pragma unroll
        for (int i = 0; i < 8; ++i) {
            out[(size_t)(rowBase + msk8 + i) * 4 + col] = acc4[i] + bias;
        }
    }
}

// ---------------------------------------------------------------------------
extern "C" void kernel_launch(void* const* d_in, const int* in_sizes, int n_in,
                              void* d_out, int out_size, void* d_ws, size_t ws_size,
                              hipStream_t stream) {
    const float* images = (const float*)d_in[0];
    const float* angles = (const float*)d_in[1];
    const float* W1 = (const float*)d_in[2];
    const float* b1 = (const float*)d_in[3];
    const float* W2 = (const float*)d_in[4];
    const float* b2 = (const float*)d_in[5];
    const float* W3 = (const float*)d_in[6];
    const float* b3 = (const float*)d_in[7];
    const float* W4 = (const float*)d_in[8];
    const float* b4 = (const float*)d_in[9];

    char* ws = (char*)d_ws;
    size_t off = 0;
    _Float16* X   = (_Float16*)(ws + off); off += (size_t)B_TOTAL * D_INF * sizeof(_Float16);
    _Float16* W1p = (_Float16*)(ws + off); off += (size_t)W1_HALVES * sizeof(_Float16);
    _Float16* W2p = (_Float16*)(ws + off); off += (size_t)W2_HALVES * sizeof(_Float16);
    _Float16* W3p = (_Float16*)(ws + off); off += (size_t)W3_HALVES * sizeof(_Float16);
    _Float16* W4p = (_Float16*)(ws + off); off += (size_t)W4_HALVES * sizeof(_Float16);
    (void)W3p; (void)W4p;

    // Pack weights into f16 WMMA B-fragment layout (W2|W3|W4 contiguous: TDM src).
    {
        int t1 = 6 * 20 * 32;
        pack_kernel<<<(t1 + 255) / 256, 256, 0, stream>>>(W1, W1p, CHN, 6, 20);
        int t2 = 6 * 3 * 32;
        pack_kernel<<<(t2 + 255) / 256, 256, 0, stream>>>(W2, W2p, CHN, 6, 3);
        pack_kernel<<<(t2 + 255) / 256, 256, 0, stream>>>(W3, W3p, CHN, 6, 3);
        pack_kernel<<<1, 96, 0, stream>>>(W4, W4p, 4, 1, 3);
    }

    // Histogram-accelerated top-200 + feature assembly (one block per image).
    feat_kernel<<<B_TOTAL, 256, 0, stream>>>(images, angles, X);

    // Fused WMMA MLP: 512 blocks x 128 threads (4 waves x 16 rows each).
    mlp_kernel<<<B_TOTAL / 64, 128, 0, stream>>>(X, W1p, W2p,
                                                 b1, b2, b3, b4, (float*)d_out);
}
